// LSTMNet_40278203302425
// MI455X (gfx1250) — matrix operand order repaired and checked by reading, not compiled, not run on hardware
//
#include <hip/hip_runtime.h>
#include <hip/hip_bf16.h>
#include <math.h>

// ---------------- problem constants ----------------
#define SEQ   2048
#define BATCH 16
#define EMB   512
#define HID   512
#define VOCAB 32000
#define NWG   16          // workgroups in the persistent recurrent kernel
#define WGT   256         // threads per WG (8 wave32s)

typedef __attribute__((ext_vector_type(16))) __bf16 v16bf;
typedef __attribute__((ext_vector_type(8)))  float  v8f;

union V16U { uint4 q[2]; v16bf v; };

static __device__ __forceinline__ unsigned short f32_to_bf16_rne(float f) {
    unsigned int u  = __float_as_uint(f);
    unsigned int lsb = (u >> 16) & 1u;
    u += 0x7fffu + lsb;                 // round-to-nearest-even
    return (unsigned short)(u >> 16);
}

// A-matrix tile (16x32 bf16, M=row, K): lanes 0-15 hold K = k0+kh*8..+7 and
// k0+16+kh*8..+7 (kh = lane>>4), per ISA 7.12.2 "16-bit A-Matrix 16x32".
static __device__ __forceinline__ v16bf load_A_tile(const unsigned short* base,
                                                    int ldK, int m, int kh, int k0) {
    const unsigned short* p = base + (size_t)m * ldK + k0 + kh * 8;
    V16U u;
    u.q[0] = *(const uint4*)(p);        // K = k0+kh*8   .. +7
    u.q[1] = *(const uint4*)(p + 16);   // K = k0+16+kh*8.. +7
    return u.v;
}

// B-matrix tile (32x16 bf16, K x N), B[k][n] = W[n0+n][k0+k]:
// lane holds fixed N = lane&15, K = k0+kh*8..+7 and k0+16+kh*8..+7 (kh = lane>>4).
static __device__ __forceinline__ v16bf load_B_tile(const unsigned short* Wrow,
                                                    int kh, int k0) {
    const unsigned short* p = Wrow + k0 + kh * 8;
    V16U u;
    u.q[0] = *(const uint4*)(p);
    u.q[1] = *(const uint4*)(p + 16);
    return u.v;
}

// ---------------- fp32 -> bf16 convert ----------------
__global__ void __launch_bounds__(256)
cvt_f32_bf16(const float* __restrict__ src, unsigned short* __restrict__ dst, int n) {
    int i = blockIdx.x * 256 + threadIdx.x;
    if (i < n) dst[i] = f32_to_bf16_rne(src[i]);
}

// ---------------- persistent LSTM recurrence ----------------
// Grid: NWG x WGT. WG k owns hidden cols [32k, 32k+32). Wave w (0..7):
// gate type g = w>>1 (i,f,g,o), sub-tile s = w&1 -> gate columns
// g*512 + 32k + 16s + [0,16). Each step: fused x_t*W_ih^T + h*W_hh^T via
// 32 chained v_wmma_f32_16x16x32_bf16, gates->LDS, LSTM cell elementwise,
// h ping-pong in global bf16, device-scope atomic barrier between steps.
__global__ void __launch_bounds__(WGT, 1)
lstm_recurrent(const unsigned short* __restrict__ sentB,   // [SEQ*16*512] bf16
               const unsigned short* __restrict__ WihB,    // [2048*512]  bf16
               const unsigned short* __restrict__ WhhB,    // [2048*512]  bf16
               const float* __restrict__ b_ih,
               const float* __restrict__ b_hh,
               unsigned short* __restrict__ hB,            // [2][16*512] bf16 ping-pong
               unsigned int* __restrict__ bar)
{
    __shared__ float gatesLds[4][16][32];   // [gate][batch][hidden-in-slice]
    __shared__ float cLds[16][32];          // cell state tile (persists all steps)
    __shared__ float bsum[4][32];           // b_ih + b_hh for this slice

    const int tid  = threadIdx.x;
    const int lane = tid & 31;
    const int wave = tid >> 5;              // 0..7
    const int g    = wave >> 1;             // gate type
    const int sub  = wave & 1;              // which 16-wide half of the 32 slice
    const int m    = lane & 15;             // batch row (A/C M index)
    const int kh   = lane >> 4;             // K-half selector
    const int hs   = blockIdx.x * 32;       // hidden slice base
    const int nlane = g * HID + hs + sub * 16 + (lane & 15);  // gate column of this lane

    for (int e = tid; e < 16 * 32; e += WGT) cLds[e >> 5][e & 31] = 0.0f;
    for (int e = tid; e < 4 * 32; e += WGT) {
        int gg = e >> 5, j = e & 31;
        int col = gg * HID + hs + j;
        bsum[gg][j] = b_ih[col] + b_hh[col];
    }
    __syncthreads();

    const unsigned short* WihRow = WihB + (size_t)nlane * EMB;
    const unsigned short* WhhRow = WhhB + (size_t)nlane * HID;

    for (int t = 0; t < SEQ; ++t) {
        const unsigned short* xrow  = sentB + (size_t)t * (BATCH * EMB);
        const unsigned short* hRead = hB + (size_t)(t & 1) * (BATCH * HID);
        unsigned short*       hWrit = hB + (size_t)((t + 1) & 1) * (BATCH * HID);

        v8f acc = {0.f, 0.f, 0.f, 0.f, 0.f, 0.f, 0.f, 0.f};

        // x_t @ W_ih^T (fused input projection, K = EMB)
        #pragma unroll
        for (int kc = 0; kc < EMB / 32; ++kc) {
            v16bf a = load_A_tile(xrow, EMB, m, kh, kc * 32);
            v16bf b = load_B_tile(WihRow, kh, kc * 32);
            acc = __builtin_amdgcn_wmma_f32_16x16x32_bf16(
                false, a, false, b, (short)0, acc, false, false);
        }
        // h_{t-1} @ W_hh^T (K = HID)
        #pragma unroll
        for (int kc = 0; kc < HID / 32; ++kc) {
            v16bf a = load_A_tile(hRead, HID, m, kh, kc * 32);
            v16bf b = load_B_tile(WhhRow, kh, kc * 32);
            acc = __builtin_amdgcn_wmma_f32_16x16x32_bf16(
                false, a, false, b, (short)0, acc, false, false);
        }

        // C/D layout: VGPR r, lane l -> M = r + 8*(l>>4), N = l&15
        #pragma unroll
        for (int r = 0; r < 8; ++r)
            gatesLds[g][r + 8 * kh][sub * 16 + (lane & 15)] = acc[r];
        __syncthreads();

        // LSTM cell for this WG's 16x32 tile (fp32 state, PyTorch gate order)
        for (int e = tid; e < 16 * 32; e += WGT) {
            int mm = e >> 5, j = e & 31;
            float gi = gatesLds[0][mm][j] + bsum[0][j];
            float gf = gatesLds[1][mm][j] + bsum[1][j];
            float gc = gatesLds[2][mm][j] + bsum[2][j];
            float go = gatesLds[3][mm][j] + bsum[3][j];
            float iv = 1.f / (1.f + __expf(-gi));
            float fv = 1.f / (1.f + __expf(-gf));
            float cv = tanhf(gc);
            float ov = 1.f / (1.f + __expf(-go));
            float cn = fv * cLds[mm][j] + iv * cv;
            cLds[mm][j] = cn;
            hWrit[mm * HID + hs + j] = f32_to_bf16_rne(ov * tanhf(cn));
        }

        // prefetch next timestep's activations while we wait at the barrier
        if (t + 1 < SEQ)
            __builtin_prefetch(sentB + (size_t)(t + 1) * (BATCH * EMB) + tid * 32, 0, 3);

        // device-scope step barrier (release h writes, acquire peers' writes)
        __threadfence();
        __syncthreads();
        if (tid == 0) {
            __hip_atomic_fetch_add(bar, 1u, __ATOMIC_ACQ_REL, __HIP_MEMORY_SCOPE_AGENT);
            const unsigned int target = (unsigned int)NWG * (unsigned int)(t + 1);
            while (__hip_atomic_load(bar, __ATOMIC_ACQUIRE, __HIP_MEMORY_SCOPE_AGENT) < target)
                __builtin_amdgcn_s_sleep(1);
        }
        __syncthreads();
    }
}

// ---------------- output projection: logits = h @ W_out^T + b_out ----------
__global__ void __launch_bounds__(256)
out_proj(const unsigned short* __restrict__ hB,     // final h, [16*512] bf16
         const unsigned short* __restrict__ WoutB,  // [32000*512] bf16
         const float* __restrict__ b_out,
         float* __restrict__ logits)                // [16*32000] f32
{
    const int lane  = threadIdx.x & 31;
    const int wave  = threadIdx.x >> 5;
    const int n0    = (blockIdx.x * 8 + wave) * 16;   // 2000 tiles over V
    const int m     = lane & 15;
    const int kh    = lane >> 4;
    const int nlane = n0 + (lane & 15);
    const unsigned short* Wrow = WoutB + (size_t)nlane * HID;

    v8f acc = {0.f, 0.f, 0.f, 0.f, 0.f, 0.f, 0.f, 0.f};
    #pragma unroll
    for (int kc = 0; kc < HID / 32; ++kc) {
        v16bf a = load_A_tile(hB, HID, m, kh, kc * 32);
        v16bf b = load_B_tile(Wrow, kh, kc * 32);
        acc = __builtin_amdgcn_wmma_f32_16x16x32_bf16(
            false, a, false, b, (short)0, acc, false, false);
    }
    float bias = b_out[nlane];
    #pragma unroll
    for (int r = 0; r < 8; ++r) {
        int mm = r + 8 * kh;
        logits[(size_t)mm * VOCAB + nlane] = acc[r] + bias;
    }
}

// ---------------- log-softmax over V per batch row ----------------
__global__ void __launch_bounds__(256)
log_softmax_rows(const float* __restrict__ logits, float* __restrict__ out)
{
    __shared__ float red[256];
    const int b = blockIdx.x;
    const float* row = logits + (size_t)b * VOCAB;

    float mx = -INFINITY;
    for (int v = threadIdx.x; v < VOCAB; v += 256) mx = fmaxf(mx, row[v]);
    red[threadIdx.x] = mx; __syncthreads();
    for (int s = 128; s > 0; s >>= 1) {
        if (threadIdx.x < s) red[threadIdx.x] = fmaxf(red[threadIdx.x], red[threadIdx.x + s]);
        __syncthreads();
    }
    mx = red[0]; __syncthreads();

    float sum = 0.f;
    for (int v = threadIdx.x; v < VOCAB; v += 256) sum += expf(row[v] - mx);
    red[threadIdx.x] = sum; __syncthreads();
    for (int s = 128; s > 0; s >>= 1) {
        if (threadIdx.x < s) red[threadIdx.x] += red[threadIdx.x + s];
        __syncthreads();
    }
    float lse = mx + logf(red[0]);

    for (int v = threadIdx.x; v < VOCAB; v += 256)
        out[(size_t)b * VOCAB + v] = row[v] - lse;
}

// ---------------- host side ----------------
extern "C" void kernel_launch(void* const* d_in, const int* in_sizes, int n_in,
                              void* d_out, int out_size, void* d_ws, size_t ws_size,
                              hipStream_t stream)
{
    (void)in_sizes; (void)n_in; (void)out_size; (void)ws_size;
    const float* sentences = (const float*)d_in[0];
    const float* W_ih      = (const float*)d_in[1];
    const float* W_hh      = (const float*)d_in[2];
    const float* b_ih      = (const float*)d_in[3];
    const float* b_hh      = (const float*)d_in[4];
    const float* W_out     = (const float*)d_in[5];
    const float* b_out     = (const float*)d_in[6];
    float* out = (float*)d_out;

    // workspace carve-up (256B aligned)
    char*  ws  = (char*)d_ws;
    size_t off = 0;
    auto carve = [&](size_t bytes) -> void* {
        void* p = ws + off;
        off = (off + bytes + 255) & ~(size_t)255;
        return p;
    };
    unsigned int*   bar   = (unsigned int*)  carve(256);
    unsigned short* hB    = (unsigned short*)carve((size_t)2 * BATCH * HID * 2); // ping-pong
    const size_t initBytes = off;                        // bar + hB need zeroing
    unsigned short* WihB  = (unsigned short*)carve((size_t)4 * HID * EMB * 2);
    unsigned short* WhhB  = (unsigned short*)carve((size_t)4 * HID * HID * 2);
    unsigned short* WoutB = (unsigned short*)carve((size_t)VOCAB * HID * 2);
    unsigned short* sentB = (unsigned short*)carve((size_t)SEQ * BATCH * EMB * 2);
    float*          logit = (float*)         carve((size_t)BATCH * VOCAB * 4);

    // deterministic per-call init (graph-capturable)
    hipMemsetAsync(d_ws, 0, initBytes, stream);

    // fp32 -> bf16 weight/activation packs
    int n;
    n = 4 * HID * EMB;       cvt_f32_bf16<<<(n + 255) / 256, 256, 0, stream>>>(W_ih,      WihB,  n);
    n = 4 * HID * HID;       cvt_f32_bf16<<<(n + 255) / 256, 256, 0, stream>>>(W_hh,      WhhB,  n);
    n = VOCAB * HID;         cvt_f32_bf16<<<(n + 255) / 256, 256, 0, stream>>>(W_out,     WoutB, n);
    n = SEQ * BATCH * EMB;   cvt_f32_bf16<<<(n + 255) / 256, 256, 0, stream>>>(sentences, sentB, n);

    // 2048-step recurrence (persistent, 16 WGs, atomic step barrier).
    // SEQ even -> final h lands in ping-pong buffer 0.
    lstm_recurrent<<<NWG, WGT, 0, stream>>>(sentB, WihB, WhhB, b_ih, b_hh, hB, bar);

    // logits + log-softmax
    out_proj<<<(VOCAB / 16) / 8, 256, 0, stream>>>(hB, WoutB, b_out, logit);
    log_softmax_rows<<<BATCH, 256, 0, stream>>>(logit, out);
}
